// EdgePoolEncoder_84344567759558
// MI455X (gfx1250) — matrix operand order repaired and checked by reading, not compiled, not run on hardware
//
// EdgePoolEncoder for MI455X (gfx1250). All dense matmuls + attention run on
// v_wmma_f32_16x16x32_f16 (fp32 data -> f16 operands, fp32 accumulate).
// ~15 GFLOP over ~50MB => latency bound on MI455X (23.3 TB/s HBM, 192MB L2).
// GEMM fast path: one wave computes a 64x16 tile (4 M-subtiles) so the
// strided B fragment is loaded once per K-step and reused by 4 WMMAs; A-side
// loads are float4 (b128) on the two contiguous 8-float runs of the
// A-fragment layout. The guarded path (only the K=13 input projection) uses
// MT=1 to avoid address-register pressure serializing its loads on xcnt.
// All fragment builds are branch-free (clamp+select). Flash attention uses a
// Bc=32 key block (2 QK^T WMMAs + dense P*V fragments), online softmax via
// lane shuffles, P-tile transposed through LDS. Segment ops use f32 global
// atomics; top-k is a deterministic global bitonic sort.
#include <hip/hip_runtime.h>
#include <hip/hip_bf16.h>
#include <math.h>

typedef __attribute__((ext_vector_type(16))) _Float16 v16h;
typedef __attribute__((ext_vector_type(8)))  float    v8f;

#define PG 5
#define NN 2048
#define EE 65536
#define CD 128
#define FF 512
#define ENS (EE + NN)          // edges + self loops
#define KEDGES 32768
#define PN (PG * NN)

// ---- output layout (floats), reference tuple concatenated flat ----
#define OUT_O0 0u          // z            (262144)
#define OUT_O1 262144u     // ei_kept      (65536)
#define OUT_O2 327680u     // escore       (65536)
#define OUT_O3 393216u     // top_idx      (32768)
#define OUT_O4 425984u     // ei_kept      (65536)
#define OUT_O5 491520u     // z            (262144)
#define OUT_O6 753664u     // all_rep      (1310720)

// A-fragment (16x32 f16) per-lane K offset for half index hi (ISA 7.12.2):
// halves 0..7  -> K = 8*grp + hi ;  halves 8..15 -> K = 16 + 8*grp + (hi-8)
__device__ __forceinline__ int a_koff(int hi, int grp) {
  return (hi < 8) ? (8 * grp + hi) : (8 + 8 * grp + hi);
}

__device__ __forceinline__ float gelu_f(float x) {
  float x3 = x * x * x;
  return 0.5f * x * (1.0f + tanhf(0.7978845608028654f * (x + 0.044715f * x3)));
}

// ---------------- WMMA GEMM: Out = act(A@B [+bias] [+Cin]) ----------------
// One wave per (16*MTv)x16 output tile (MTv M-subtiles reuse one B fragment).
// A: MxK row-major (lda), B: KxN row-major. M must be a multiple of 16*MTv
// and N of 16 (true at every call site). GUARDK=1 handles K%32!=0 branch-free.
template <int ACT, int BIAS_EN, int ADD_EN, int GUARDK, int MTv>
__global__ void gemm_wmma(const float* __restrict__ A, int lda,
                          const float* __restrict__ Bm, int ldb,
                          const float* __restrict__ bias,
                          const float* __restrict__ Cin, int ldc,
                          float* __restrict__ Out, int ldo,
                          int N, int K) {
  int lane = threadIdx.x;
  int grp = lane >> 4, lr = lane & 15;
  int m0 = blockIdx.y * (16 * MTv), n0 = blockIdx.x * 16;
  int bn = n0 + lr;
  v8f acc[MTv];
#pragma unroll
  for (int mt = 0; mt < MTv; ++mt)
#pragma unroll
    for (int v = 0; v < 8; ++v) acc[mt][v] = 0.0f;

  for (int k0 = 0; k0 < K; k0 += 32) {
    // ---- B fragment (built once, reused by MTv WMMAs) ----
    v16h b;
    if (GUARDK) {
      float bf[16];
#pragma unroll
      for (int hi = 0; hi < 16; ++hi) {
        int kb = k0 + a_koff(hi, grp);
        int kbc = (kb < K) ? kb : (K - 1);
        bf[hi] = Bm[(size_t)kbc * ldb + bn];
        bf[hi] = (kb < K) ? bf[hi] : 0.0f;
      }
#pragma unroll
      for (int hi = 0; hi < 16; ++hi) b[hi] = (_Float16)bf[hi];
    } else {
      float bf[16];
#pragma unroll
      for (int hi = 0; hi < 16; ++hi)
        bf[hi] = Bm[(size_t)(k0 + a_koff(hi, grp)) * ldb + bn];
#pragma unroll
      for (int hi = 0; hi < 16; ++hi) b[hi] = (_Float16)bf[hi];
    }
    // ---- MTv A fragments + WMMAs ----
#pragma unroll
    for (int mt = 0; mt < MTv; ++mt) {
      int am = m0 + mt * 16 + lr;
      v16h a;
      if (GUARDK) {
        float af[16];
#pragma unroll
        for (int hi = 0; hi < 16; ++hi) {
          int ka = k0 + a_koff(hi, grp);
          int kac = (ka < K) ? ka : (K - 1);
          af[hi] = A[(size_t)am * lda + kac];
          af[hi] = (ka < K) ? af[hi] : 0.0f;
        }
#pragma unroll
        for (int hi = 0; hi < 16; ++hi) a[hi] = (_Float16)af[hi];
      } else {
        const float* arow = A + (size_t)am * lda + k0 + 8 * grp;
        const float4* av4 = (const float4*)arow;
        float4 a0 = av4[0], a1 = av4[1], a2 = av4[4], a3 = av4[5];
        float af[16] = {a0.x, a0.y, a0.z, a0.w, a1.x, a1.y, a1.z, a1.w,
                        a2.x, a2.y, a2.z, a2.w, a3.x, a3.y, a3.z, a3.w};
#pragma unroll
        for (int hi = 0; hi < 16; ++hi) a[hi] = (_Float16)af[hi];
      }
      acc[mt] = __builtin_amdgcn_wmma_f32_16x16x32_f16(
          false, a, false, b, (short)0, acc[mt], false, false);
    }
  }
  int n = n0 + lr;
  float bval = BIAS_EN ? bias[n] : 0.0f;
#pragma unroll
  for (int mt = 0; mt < MTv; ++mt)
#pragma unroll
    for (int v = 0; v < 8; ++v) {
      int m = m0 + mt * 16 + v + 8 * grp;
      float val = acc[mt][v] + bval;
      if (ADD_EN) val += Cin[(size_t)m * ldc + n];
      if (ACT == 1) val = 1.0f / (1.0f + expf(-val));
      else if (ACT == 2) val = tanhf(val);
      else if (ACT == 3) val = gelu_f(val);
      Out[(size_t)m * ldo + n] = val;
    }
}

// --------- fused flash attention: one wave per (16-query tile, head) --------
// Q,K,V,O: [N x C], head in columns [head*DH, (head+1)*DH). grid.z = graph.
// 32 keys per iteration: 2 QK^T WMMAs + dense P*V fragments.
template <int DH>
__global__ void attn_flash(const float* __restrict__ Qm,
                           const float* __restrict__ Km,
                           const float* __restrict__ Vm,
                           float* __restrict__ Om, int N, int C) {
  __shared__ float sP[16][36];  // 16x32 P tile, 36-float row stride
  int lane = threadIdx.x, grp = lane >> 4, lr = lane & 15;
  size_t gbase = (size_t)blockIdx.z * N * C;
  const float* Q = Qm + gbase;
  const float* Kp = Km + gbase;
  const float* Vp = Vm + gbase;
  float* Op = Om + gbase;
  int hcol = blockIdx.y * DH;
  int q0 = blockIdx.x * 16;
  const float scale = rsqrtf((float)DH);

  // Q fragment (A layout). Halves 0..7: K=8*grp+i (always < DH);
  // halves 8..15: K=16+8*grp+i (valid only for DH=32).
  v16h aq;
  {
    const float* qrow = Q + (size_t)(q0 + lr) * C + hcol + 8 * grp;
    const float4* q4 = (const float4*)qrow;
    float4 r0 = q4[0], r1 = q4[1];
    float qf[16] = {r0.x, r0.y, r0.z, r0.w, r1.x, r1.y, r1.z, r1.w,
                    0, 0, 0, 0, 0, 0, 0, 0};
    if (DH == 32) {
      float4 r2 = q4[4], r3 = q4[5];
      qf[8] = r2.x;  qf[9] = r2.y;  qf[10] = r2.z; qf[11] = r2.w;
      qf[12] = r3.x; qf[13] = r3.y; qf[14] = r3.z; qf[15] = r3.w;
    }
#pragma unroll
    for (int hi = 0; hi < 16; ++hi) aq[hi] = (_Float16)qf[hi];
  }

  float mrow[8], lrow[8];
  v8f Oacc[DH / 16];
#pragma unroll
  for (int v = 0; v < 8; ++v) { mrow[v] = -3.0e38f; lrow[v] = 0.0f; }
#pragma unroll
  for (int j = 0; j < DH / 16; ++j)
#pragma unroll
    for (int v = 0; v < 8; ++v) Oacc[j][v] = 0.0f;

  for (int kt = 0; kt < N / 32; ++kt) {
    int kb0 = kt * 32;
    // ---- S = Q @ K^T for 32 keys (two 16-key WMMAs) ----
    v8f s01[2];
#pragma unroll
    for (int t2 = 0; t2 < 2; ++t2) {
      // B fragment: halves hi -> K-dim = a_koff(hi, grp), N = key = lr.
      const float* krow = Kp + (size_t)(kb0 + 16 * t2 + lr) * C + hcol;
      int koff = 8 * grp;  // DH=16: halves 8..15 zeroed below
      const float4* k4 = (const float4*)(krow + koff);
      float4 r0 = k4[0], r1 = k4[1], r2 = k4[(DH == 32) ? 4 : 0], r3 = k4[(DH == 32) ? 5 : 1];
      float kf[16] = {r0.x, r0.y, r0.z, r0.w, r1.x, r1.y, r1.z, r1.w,
                      r2.x, r2.y, r2.z, r2.w, r3.x, r3.y, r3.z, r3.w};
      v16h bk;
#pragma unroll
      for (int hi = 0; hi < 16; ++hi) {
        float kv = kf[hi];
        if (DH == 16) kv = (hi < 8) ? kv : 0.0f;
        bk[hi] = (_Float16)kv;
      }
      v8f z8;
#pragma unroll
      for (int v = 0; v < 8; ++v) z8[v] = 0.0f;
      s01[t2] = __builtin_amdgcn_wmma_f32_16x16x32_f16(
          false, aq, false, bk, (short)0, z8, false, false);
    }
    // ---- online softmax over the 32-key block ----
    float p0[8], p1[8];
#pragma unroll
    for (int v = 0; v < 8; ++v) {
      float sv0 = s01[0][v] * scale;
      float sv1 = s01[1][v] * scale;
      float rmax = fmaxf(sv0, sv1);
      for (int off = 1; off < 16; off <<= 1)
        rmax = fmaxf(rmax, __shfl_xor(rmax, off, 32));
      float mnew = fmaxf(mrow[v], rmax);
      float fs = expf(mrow[v] - mnew);
      float e0 = expf(sv0 - mnew), e1 = expf(sv1 - mnew);
      float rsum = e0 + e1;
      for (int off = 1; off < 16; off <<= 1) rsum += __shfl_xor(rsum, off, 32);
      lrow[v] = lrow[v] * fs + rsum;
      mrow[v] = mnew;
#pragma unroll
      for (int j = 0; j < DH / 16; ++j) Oacc[j][v] *= fs;
      p0[v] = e0;
      p1[v] = e1;
    }
    // ---- transpose P (16x32) through LDS into A-fragment layout ----
#pragma unroll
    for (int v = 0; v < 8; ++v) {
      sP[v + 8 * grp][lr] = p0[v];
      sP[v + 8 * grp][16 + lr] = p1[v];
    }
    __syncthreads();
    v16h ap;
    {
      const float* prow = &sP[lr][0];
      const float4* p4a = (const float4*)(prow + 8 * grp);
      const float4* p4b = (const float4*)(prow + 16 + 8 * grp);
      float4 r0 = p4a[0], r1 = p4a[1], r2 = p4b[0], r3 = p4b[1];
      float pf[16] = {r0.x, r0.y, r0.z, r0.w, r1.x, r1.y, r1.z, r1.w,
                      r2.x, r2.y, r2.z, r2.w, r3.x, r3.y, r3.z, r3.w};
#pragma unroll
      for (int hi = 0; hi < 16; ++hi) ap[hi] = (_Float16)pf[hi];
    }
    __syncthreads();
    // ---- O += P @ V (dense: 32 keys in K-dim) ----
#pragma unroll
    for (int j = 0; j < DH / 16; ++j) {
      v16h bv;
#pragma unroll
      for (int hi = 0; hi < 16; ++hi) {
        int kk = a_koff(hi, grp);  // key within 32-block
        bv[hi] =
            (_Float16)Vp[(size_t)(kb0 + kk) * C + hcol + j * 16 + lr];
      }
      Oacc[j] = __builtin_amdgcn_wmma_f32_16x16x32_f16(
          false, ap, false, bv, (short)0, Oacc[j], false, false);
    }
  }
#pragma unroll
  for (int j = 0; j < DH / 16; ++j)
#pragma unroll
    for (int v = 0; v < 8; ++v) {
      int m = q0 + v + 8 * grp;
      Op[(size_t)m * C + hcol + j * 16 + lr] =
          Oacc[j][v] / (lrow[v] + 1e-16f);
    }
}

// ---------------------------- small kernels -------------------------------
__global__ void k_fill0(float* p, int n) {
  int i = blockIdx.x * blockDim.x + threadIdx.x;
  if (i < n) p[i] = 0.0f;
}
__global__ void k_fill_bias_rows(float* p, const float* b, int n) {
  int i = blockIdx.x * blockDim.x + threadIdx.x;
  if (i < n) p[i] = b[i & (CD - 1)];
}
__global__ void k_eadd(float* a, const float* b, int n) {
  int i = blockIdx.x * blockDim.x + threadIdx.x;
  if (i < n) a[i] += b[i];
}
__global__ void k_emul(float* o, const float* a, const float* b, int n) {
  int i = blockIdx.x * blockDim.x + threadIdx.x;
  if (i < n) o[i] = a[i] * b[i];
}
__global__ void k_copy(float* o, const float* a, int n) {
  int i = blockIdx.x * blockDim.x + threadIdx.x;
  if (i < n) o[i] = a[i];
}
__global__ void k_add_temb(float* h, const float* temb, int n) {
  int i = blockIdx.x * blockDim.x + threadIdx.x;
  if (i < n) h[i] += temb[(i / (NN * CD)) * CD + (i & (CD - 1))];
}
__global__ void k_posdiff(const float* X, const int* ei, float* pd, int g) {
  int e = blockIdx.x * blockDim.x + threadIdx.x;
  if (e >= EE) return;
  int s = ei[e], d = ei[EE + e];
  const float* xs = X + (size_t)(g * NN + s) * 16;
  const float* xd = X + (size_t)(g * NN + d) * 16;
  pd[e * 3 + 0] = xs[0] - xd[0];
  pd[e * 3 + 1] = xs[1] - xd[1];
  pd[e * 3 + 2] = xs[2] - xd[2];
}
__device__ __forceinline__ float pt_delta(const float* pd, const float* Wp,
                                          const float* bp, int e, int c) {
  return pd[e * 3 + 0] * Wp[c] + pd[e * 3 + 1] * Wp[CD + c] +
         pd[e * 3 + 2] * Wp[2 * CD + c] + bp[c];
}
__global__ void k_pt_pass1(const float* qb, const float* kb, const float* pd,
                           const float* Wp, const float* bp, const int* ei,
                           float* ssum) {
  int i = blockIdx.x * blockDim.x + threadIdx.x;
  if (i >= EE * CD) return;
  int e = i >> 7, c = i & 127;
  int s = ei[e], d = ei[EE + e];
  float delta = pt_delta(pd, Wp, bp, e, c);
  float logit = qb[d * CD + c] - kb[s * CD + c] + delta;
  atomicAdd(&ssum[d * CD + c], expf(logit));
}
__global__ void k_pt_pass2(const float* qb, const float* kb, const float* vb,
                           const float* pd, const float* Wp, const float* bp,
                           const int* ei, const float* ssum, float* dr) {
  int i = blockIdx.x * blockDim.x + threadIdx.x;
  if (i >= EE * CD) return;
  int e = i >> 7, c = i & 127;
  int s = ei[e], d = ei[EE + e];
  float delta = pt_delta(pd, Wp, bp, e, c);
  float logit = qb[d * CD + c] - kb[s * CD + c] + delta;
  float alpha = expf(logit) / (ssum[d * CD + c] + 1e-16f);
  atomicAdd(&dr[d * CD + c], alpha * (vb[s * CD + c] + delta));
}
__global__ void k_deg(const int* ei, float* deg) {
  int e = blockIdx.x * blockDim.x + threadIdx.x;
  if (e >= ENS) return;
  int d = (e < EE) ? ei[EE + e] : (e - EE);
  atomicAdd(&deg[d], 1.0f);
}
__global__ void k_norm(const int* ei, const float* deg, float* nrm) {
  int e = blockIdx.x * blockDim.x + threadIdx.x;
  if (e >= ENS) return;
  int s = (e < EE) ? ei[e] : (e - EE);
  int d = (e < EE) ? ei[EE + e] : (e - EE);
  nrm[e] = rsqrtf(deg[s] * deg[d]);
}
__global__ void k_scatter(const float* tmp, const int* ei, const float* nrm,
                          float* accg) {
  int i = blockIdx.x * blockDim.x + threadIdx.x;
  if (i >= ENS * CD) return;
  int e = i >> 7, c = i & 127;
  int s = (e < EE) ? ei[e] : (e - EE);
  int d = (e < EE) ? ei[EE + e] : (e - EE);
  atomicAdd(&accg[d * CD + c], tmp[s * CD + c] * nrm[e]);
}
__global__ void k_softmax5(const float* att, float* probs) {
  if (threadIdx.x == 0 && blockIdx.x == 0) {
    float m = -3.0e38f;
    for (int p = 0; p < PG; ++p) m = fmaxf(m, att[p]);
    float s = 0.0f, e[PG];
    for (int p = 0; p < PG; ++p) { e[p] = expf(att[p] - m); s += e[p]; }
    for (int p = 0; p < PG; ++p) probs[p] = e[p] / s;
  }
}
__global__ void k_gru(float* Hs, float* accz, const float* Z, const float* Ht,
                      const float* probs, int t, int n) {
  int i = blockIdx.x * blockDim.x + threadIdx.x;
  if (i >= n) return;
  float z = Z[i];
  float h = z * Hs[i] + (1.0f - z) * Ht[i];
  Hs[i] = h;
  accz[i] += probs[t] * h;
}
__global__ void k_layernorm(const float* x, const float* g, const float* b,
                            float* o) {
  int row = blockIdx.x, lane = threadIdx.x;
  float v[4];
  float s = 0.0f;
#pragma unroll
  for (int i = 0; i < 4; ++i) { v[i] = x[row * CD + lane + 32 * i]; s += v[i]; }
  for (int off = 16; off > 0; off >>= 1) s += __shfl_xor(s, off, 32);
  float mu = s * (1.0f / CD);
  float var = 0.0f;
#pragma unroll
  for (int i = 0; i < 4; ++i) { float d = v[i] - mu; var += d * d; }
  for (int off = 16; off > 0; off >>= 1) var += __shfl_xor(var, off, 32);
  var *= (1.0f / CD);
  float inv = rsqrtf(var + 1e-5f);
#pragma unroll
  for (int i = 0; i < 4; ++i) {
    int c = lane + 32 * i;
    o[row * CD + c] = (v[i] - mu) * inv * g[c] + b[c];
  }
}
__global__ void k_escore(const float* z, const int* ei, const float* Ws,
                         const float* bs, float* esc) {
  int e = blockIdx.x * blockDim.x + threadIdx.x;
  if (e >= EE) return;
  int s = ei[e], d = ei[EE + e];
  float acc = 0.0f;
  for (int c = 0; c < CD; ++c)
    acc += z[s * CD + c] * Ws[c] + z[d * CD + c] * Ws[CD + c];
  esc[e] = 1.0f / (1.0f + expf(-(acc + bs[0])));
}
__global__ void k_sortinit(int* idx) {
  int e = blockIdx.x * blockDim.x + threadIdx.x;
  if (e < EE) idx[e] = e;
}
__global__ void k_bitonic(float* key, int* idx, int size, int stride) {
  int i = blockIdx.x * blockDim.x + threadIdx.x;
  if (i >= EE) return;
  int p = i ^ stride;
  if (p <= i) return;
  bool descBlock = ((i & size) == 0);
  float ki = key[i], kp = key[p];
  bool sw = descBlock ? (ki < kp) : (ki > kp);
  if (sw) {
    key[i] = kp; key[p] = ki;
    int t = idx[i]; idx[i] = idx[p]; idx[p] = t;
  }
}
__global__ void k_emit_topk(const int* idx, const int* ei, float* out) {
  int j = blockIdx.x * blockDim.x + threadIdx.x;
  if (j >= KEDGES) return;
  int t = idx[j];
  float s = (float)ei[t], d = (float)ei[EE + t];
  out[OUT_O1 + j] = s;
  out[OUT_O1 + KEDGES + j] = d;
  out[OUT_O3 + j] = (float)t;
  out[OUT_O4 + j] = s;
  out[OUT_O4 + KEDGES + j] = d;
}

// ---------------------------- host helpers --------------------------------
static void gemm(hipStream_t st, const float* A, int lda, const float* Bm,
                 int ldb, const float* bias, const float* Cin, int ldc,
                 float* Out, int ldo, int M, int N, int K, int act) {
  dim3 blk(32, 1, 1);
  if (K % 32 != 0) {  // only the 13-wide input projection: MT=1, low VGPR use
    dim3 grid((unsigned)(N / 16), (unsigned)(M / 16), 1);
    gemm_wmma<0, 0, 0, 1, 1><<<grid, blk, 0, st>>>(A, lda, Bm, ldb, bias, Cin,
                                                   ldc, Out, ldo, N, K);
    return;
  }
  dim3 grid((unsigned)(N / 16), (unsigned)(M / 64), 1);
  int be = bias ? 1 : 0, ae = Cin ? 1 : 0;
#define GL(ACTV, BEV, AEV)                                                    \
  gemm_wmma<ACTV, BEV, AEV, 0, 4><<<grid, blk, 0, st>>>(                      \
      A, lda, Bm, ldb, bias, Cin, ldc, Out, ldo, N, K)
  switch (act * 4 + be * 2 + ae) {
    case 0:  GL(0, 0, 0); break;
    case 1:  GL(0, 0, 1); break;
    case 2:  GL(0, 1, 0); break;
    case 3:  GL(0, 1, 1); break;
    case 4:  GL(1, 0, 0); break;
    case 5:  GL(1, 0, 1); break;
    case 6:  GL(1, 1, 0); break;
    case 7:  GL(1, 1, 1); break;
    case 8:  GL(2, 0, 0); break;
    case 9:  GL(2, 0, 1); break;
    case 10: GL(2, 1, 0); break;
    case 11: GL(2, 1, 1); break;
    case 12: GL(3, 0, 0); break;
    case 13: GL(3, 0, 1); break;
    case 14: GL(3, 1, 0); break;
    case 15: GL(3, 1, 1); break;
  }
#undef GL
}

static inline dim3 g1d(int n) { return dim3((unsigned)((n + 255) / 256)); }

extern "C" void kernel_launch(void* const* d_in, const int* in_sizes, int n_in,
                              void* d_out, int out_size, void* d_ws,
                              size_t ws_size, hipStream_t stream) {
  (void)in_sizes; (void)n_in; (void)out_size; (void)ws_size;
  // ---- inputs (setup_inputs dict order, recursive) ----
  const float* X      = (const float*)d_in[0];
  const int*   EI     = (const int*)d_in[1];
  const float* W_in   = (const float*)d_in[2];
  const float* W_lin  = (const float*)d_in[3];
  const float* W_srcM = (const float*)d_in[4];
  const float* W_dstM = (const float*)d_in[5];
  const float* W_posM = (const float*)d_in[6];
  const float* b_pos  = (const float*)d_in[7];
  const float* saWq = (const float*)d_in[8];
  const float* saWk = (const float*)d_in[9];
  const float* saWv = (const float*)d_in[10];
  const float* saWo = (const float*)d_in[11];
  const float* saW1 = (const float*)d_in[12];
  const float* sab1 = (const float*)d_in[13];
  const float* saW2 = (const float*)d_in[14];
  const float* sab2 = (const float*)d_in[15];
  const float* temb = (const float*)d_in[16];
  const float* W_score = (const float*)d_in[17];
  const float* b_score = (const float*)d_in[18];
  const float* a3att = (const float*)d_in[19];
  // a3 gates: z @20..23, r @24..27, h @28..31 : Wg, bg, Wl, bl
  float* out = (float*)d_out;

  // ---- workspace layout (floats) ----
  const size_t NC = (size_t)NN * CD, PNC = (size_t)PG * NN * CD,
               NF = (size_t)NN * FF;
  float* ws = (float*)d_ws;
  float* H      = ws;                    // PNC
  float* ALLREP = H + PNC;               // PNC
  float* QB     = ALLREP + PNC;          // NC
  float* KB     = QB + NC;
  float* VB     = KB + NC;
  float* OB     = VB + NC;
  float* AB     = OB + NC;
  float* F1     = AB + NC;               // NF
  float* PD     = F1 + NF;               // 3*EE
  float* SSUM   = PD + 3 * (size_t)EE;   // NC
  float* TMP    = SSUM + NC;
  float* GZ     = TMP + NC;
  float* GR     = GZ + NC;
  float* GH     = GR + NC;
  float* T1     = GH + NC;
  float* ZB     = T1 + NC;
  float* RB     = ZB + NC;
  float* HTB    = RB + NC;
  float* HRB    = HTB + NC;
  float* HS     = HRB + NC;
  float* ZACC   = HS + NC;
  float* ZN     = ZACC + NC;
  float* DEG    = ZN + NC;               // NN
  float* NRM    = DEG + NN;              // ENS
  float* PROBS  = NRM + ENS;             // 8
  float* KEYS   = PROBS + 8;             // EE
  int*   IDXB   = (int*)(KEYS + EE);     // EE

  // 1) h = feat(:, :13) @ W_in   (x row stride 16; K=13 -> guarded path)
  gemm(stream, X, 16, W_in, CD, nullptr, nullptr, 0, H, CD, PN, CD, 13, 0);

  // 2) point transformer per graph -> dr = all_rep; h += dr
  k_fill0<<<g1d((int)PNC), 256, 0, stream>>>(ALLREP, (int)PNC);
  for (int g = 0; g < PG; ++g) {
    float* hg = H + (size_t)g * NC;
    float* drg = ALLREP + (size_t)g * NC;
    gemm(stream, hg, CD, W_dstM, CD, nullptr, nullptr, 0, QB, CD, NN, CD, CD, 0);
    gemm(stream, hg, CD, W_srcM, CD, nullptr, nullptr, 0, KB, CD, NN, CD, CD, 0);
    gemm(stream, hg, CD, W_lin,  CD, nullptr, nullptr, 0, VB, CD, NN, CD, CD, 0);
    k_posdiff<<<g1d(EE), 256, 0, stream>>>(X, EI, PD, g);
    k_fill0<<<g1d((int)NC), 256, 0, stream>>>(SSUM, (int)NC);
    k_pt_pass1<<<g1d(EE * CD), 256, 0, stream>>>(QB, KB, PD, W_posM, b_pos, EI, SSUM);
    k_pt_pass2<<<g1d(EE * CD), 256, 0, stream>>>(QB, KB, VB, PD, W_posM, b_pos, EI, SSUM, drg);
    k_eadd<<<g1d((int)NC), 256, 0, stream>>>(hg, drg, (int)NC);
  }

  // 3) self-attention (8 heads, dh=16) + FFN, residual into h
  for (int g = 0; g < PG; ++g) {
    float* hg = H + (size_t)g * NC;
    gemm(stream, hg, CD, saWq, CD, nullptr, nullptr, 0, QB, CD, NN, CD, CD, 0);
    gemm(stream, hg, CD, saWk, CD, nullptr, nullptr, 0, KB, CD, NN, CD, CD, 0);
    gemm(stream, hg, CD, saWv, CD, nullptr, nullptr, 0, VB, CD, NN, CD, CD, 0);
    attn_flash<16><<<dim3(NN / 16, 8, 1), 32, 0, stream>>>(QB, KB, VB, OB, NN, CD);
    gemm(stream, OB, CD, saWo, CD, nullptr, nullptr, 0, AB, CD, NN, CD, CD, 0);
    gemm(stream, AB, CD, saW1, FF, sab1, nullptr, 0, F1, FF, NN, FF, CD, 3); // gelu
    gemm(stream, F1, FF, saW2, CD, sab2, hg, CD, hg, CD, NN, CD, FF, 0);     // +res
  }

  // 4) h += temb
  k_add_temb<<<g1d((int)PNC), 256, 0, stream>>>(H, temb, (int)PNC);

  // 5) degree norm + time-attention probs
  k_fill0<<<g1d(NN), 256, 0, stream>>>(DEG, NN);
  k_deg<<<g1d(ENS), 256, 0, stream>>>(EI, DEG);
  k_norm<<<g1d(ENS), 256, 0, stream>>>(EI, DEG, NRM);
  k_softmax5<<<1, 32, 0, stream>>>(a3att, PROBS);

  // 6) TGCN over P steps
  k_fill0<<<g1d((int)NC), 256, 0, stream>>>(HS, (int)NC);
  k_fill0<<<g1d((int)NC), 256, 0, stream>>>(ZACC, (int)NC);
  const float* Wg[3] = {(const float*)d_in[20], (const float*)d_in[24], (const float*)d_in[28]};
  const float* bg[3] = {(const float*)d_in[21], (const float*)d_in[25], (const float*)d_in[29]};
  const float* Wl[3] = {(const float*)d_in[22], (const float*)d_in[26], (const float*)d_in[30]};
  const float* bl[3] = {(const float*)d_in[23], (const float*)d_in[27], (const float*)d_in[31]};
  float* Gbuf[3] = {GZ, GR, GH};
  for (int t = 0; t < PG; ++t) {
    float* Xt = H + (size_t)t * NC;
    for (int gi = 0; gi < 3; ++gi) {
      gemm(stream, Xt, CD, Wg[gi], CD, nullptr, nullptr, 0, TMP, CD, NN, CD, CD, 0);
      k_fill_bias_rows<<<g1d((int)NC), 256, 0, stream>>>(Gbuf[gi], bg[gi], (int)NC);
      k_scatter<<<g1d(ENS * CD), 256, 0, stream>>>(TMP, EI, NRM, Gbuf[gi]);
    }
    // Z = sigmoid(gcnZ@Wl_top + Hs@Wl_bot + bl)
    gemm(stream, GZ, CD, Wl[0], CD, nullptr, nullptr, 0, T1, CD, NN, CD, CD, 0);
    gemm(stream, HS, CD, Wl[0] + CD * CD, CD, bl[0], T1, CD, ZB, CD, NN, CD, CD, 1);
    // R
    gemm(stream, GR, CD, Wl[1], CD, nullptr, nullptr, 0, T1, CD, NN, CD, CD, 0);
    gemm(stream, HS, CD, Wl[1] + CD * CD, CD, bl[1], T1, CD, RB, CD, NN, CD, CD, 1);
    // Ht = tanh(gcnH@Wl_top + (Hs*R)@Wl_bot + bl)
    k_emul<<<g1d((int)NC), 256, 0, stream>>>(HRB, HS, RB, (int)NC);
    gemm(stream, GH, CD, Wl[2], CD, nullptr, nullptr, 0, T1, CD, NN, CD, CD, 0);
    gemm(stream, HRB, CD, Wl[2] + CD * CD, CD, bl[2], T1, CD, HTB, CD, NN, CD, CD, 2);
    k_gru<<<g1d((int)NC), 256, 0, stream>>>(HS, ZACC, ZB, HTB, PROBS, t, (int)NC);
  }

  // 7) transformer layers on z (4 heads, dh=32)
  float* z = ZACC;
  for (int l = 0; l < 2; ++l) {
    int b = 32 + l * 12;
    const float* g1  = (const float*)d_in[b + 0];
    const float* b1  = (const float*)d_in[b + 1];
    const float* Wq  = (const float*)d_in[b + 2];
    const float* Wk  = (const float*)d_in[b + 3];
    const float* Wv  = (const float*)d_in[b + 4];
    const float* Wo  = (const float*)d_in[b + 5];
    const float* g2  = (const float*)d_in[b + 6];
    const float* b2  = (const float*)d_in[b + 7];
    const float* W1  = (const float*)d_in[b + 8];
    const float* bf1 = (const float*)d_in[b + 9];
    const float* W2  = (const float*)d_in[b + 10];
    const float* bf2 = (const float*)d_in[b + 11];
    k_layernorm<<<NN, 32, 0, stream>>>(z, g1, b1, ZN);
    gemm(stream, ZN, CD, Wq, CD, nullptr, nullptr, 0, QB, CD, NN, CD, CD, 0);
    gemm(stream, ZN, CD, Wk, CD, nullptr, nullptr, 0, KB, CD, NN, CD, CD, 0);
    gemm(stream, ZN, CD, Wv, CD, nullptr, nullptr, 0, VB, CD, NN, CD, CD, 0);
    attn_flash<32><<<dim3(NN / 16, 4, 1), 32, 0, stream>>>(QB, KB, VB, OB, NN, CD);
    gemm(stream, OB, CD, Wo, CD, nullptr, z, CD, z, CD, NN, CD, CD, 0);
    k_layernorm<<<NN, 32, 0, stream>>>(z, g2, b2, ZN);
    gemm(stream, ZN, CD, W1, FF, bf1, nullptr, 0, F1, FF, NN, FF, CD, 3);
    gemm(stream, F1, FF, W2, CD, bf2, z, CD, z, CD, NN, CD, FF, 0);
  }

  // 8) edge scores, top-k via full bitonic sort (descending)
  k_escore<<<g1d(EE), 256, 0, stream>>>(z, EI, W_score, b_score, KEYS);
  k_copy<<<g1d(EE), 256, 0, stream>>>(out + OUT_O2, KEYS, EE);
  k_sortinit<<<g1d(EE), 256, 0, stream>>>(IDXB);
  for (int size = 2; size <= EE; size <<= 1)
    for (int stride = size >> 1; stride > 0; stride >>= 1)
      k_bitonic<<<g1d(EE), 256, 0, stream>>>(KEYS, IDXB, size, stride);
  k_emit_topk<<<g1d(KEDGES), 256, 0, stream>>>(IDXB, EI, out);

  // 9) remaining outputs: z (twice) + all_rep
  k_copy<<<g1d((int)NC), 256, 0, stream>>>(out + OUT_O0, z, (int)NC);
  k_copy<<<g1d((int)NC), 256, 0, stream>>>(out + OUT_O5, z, (int)NC);
  k_copy<<<g1d((int)PNC), 256, 0, stream>>>(out + OUT_O6, ALLREP, (int)PNC);
}
